// RoPETransformerEncoderLayer_6794638262321
// MI455X (gfx1250) — compile-verified, matrix operand order checked by reading, not run on hardware
//
#include <hip/hip_runtime.h>

typedef unsigned int       u32;
typedef unsigned short     u16;
typedef unsigned long long u64;
typedef __attribute__((ext_vector_type(16))) __bf16 v16bf;
typedef __attribute__((ext_vector_type(8)))  float  v8f;
typedef __attribute__((ext_vector_type(8)))  u32    v8u;
typedef __attribute__((ext_vector_type(4)))  u32    v4u;

union BU { v16bf h; v8u u; v4u q[2]; };

#define Bsz 4
#define Nseq 2048
#define Dm 256
#define Hh 8
#define dh 32
#define SCALE 0.17677669529663687f   // 1/sqrt(32)

// ---- helpers -------------------------------------------------------------

static __device__ inline u16 f2bf(float f) {
    u32 u = __float_as_uint(f);
    u32 r = (u + 0x7FFFu + ((u >> 16) & 1u)) >> 16;
    return (u16)r;
}

static __device__ inline v8f wmma_bf16(const BU& a, const BU& b, v8f c) {
    return __builtin_amdgcn_wmma_f32_16x16x32_bf16(false, a.h, false, b.h,
                                                   (short)0, c, false, false);
}

static __device__ inline float redmax16(float x) {
    x = fmaxf(x, __shfl_xor(x, 1, 32));
    x = fmaxf(x, __shfl_xor(x, 2, 32));
    x = fmaxf(x, __shfl_xor(x, 4, 32));
    x = fmaxf(x, __shfl_xor(x, 8, 32));
    return x;
}
static __device__ inline float redsum16(float x) {
    x += __shfl_xor(x, 1, 32);
    x += __shfl_xor(x, 2, 32);
    x += __shfl_xor(x, 4, 32);
    x += __shfl_xor(x, 8, 32);
    return x;
}

// load one WMMA bf16 operand (16x32 A or 32x16 B tile slice for this lane)
// as two contiguous 16B chunks: elements [e, e+8) and [e+16, e+24),
// where e = rowbase + (lane>>4)*8 (bf16 element index, multiple of 8).
static __device__ inline void ld_op_global(BU& r, const v4u* p, int e) {
    r.q[0] = p[e >> 3];
    r.q[1] = p[(e >> 3) + 2];
}

// async copy of 16 bytes global -> LDS (tracked by ASYNCcnt)
static __device__ inline void async_g2l_b128(u32 lds_byte_addr, const void* gaddr) {
    asm volatile("global_load_async_to_lds_b128 %0, %1, off"
                 :: "v"(lds_byte_addr), "v"((u64)gaddr)
                 : "memory");
}
static __device__ inline void wait_asynccnt0() {
    asm volatile("s_wait_asynccnt 0" ::: "memory");
}

// ---- weight convert + transpose to bf16 (WT[col*K + k]) ------------------

__global__ void k_transpose_bf(const float* __restrict__ W, u16* __restrict__ WT,
                               int K, int NC) {
    int idx = blockIdx.x * 256 + threadIdx.x;
    if (idx >= K * NC) return;
    int k = idx / NC, c = idx % NC;
    WT[c * K + k] = f2bf(W[idx]);
}

// ---- LayerNorm over rows of 256, output bf16 -----------------------------

__global__ void k_layernorm(const float* __restrict__ x, const float* __restrict__ g,
                            const float* __restrict__ b, u16* __restrict__ out) {
    __shared__ float red[256];
    int row = blockIdx.x, t = threadIdx.x;
    float v = x[row * Dm + t];
    red[t] = v;
    __syncthreads();
    #pragma unroll
    for (int s = 128; s > 0; s >>= 1) { if (t < s) red[t] += red[t + s]; __syncthreads(); }
    float mu = red[0] * (1.0f / Dm);
    __syncthreads();
    float dv = v - mu;
    red[t] = dv * dv;
    __syncthreads();
    #pragma unroll
    for (int s = 128; s > 0; s >>= 1) { if (t < s) red[t] += red[t + s]; __syncthreads(); }
    float var = red[0] * (1.0f / Dm);
    float y = dv * rsqrtf(var + 1e-5f) * g[t] + b[t];
    out[row * Dm + t] = f2bf(y);
}

// ---- generic WMMA GEMM ----------------------------------------------------
// Block tile: 32 rows x 256 cols, 8 waves as 2(row) x 4(col); wave tile 16x64
// (4 accumulators sharing one A operand). A staged in LDS via async DMA.
// All WMMA operands fetched as 2x b128.
// modes: 0 = proj (+bias +resid -> f32)
//        1 = ffn1 (+bias, relu -> bf16)
//        2 = ffn2 (+bias +resid -> f32)
//        3 = qkv  (+bias, RoPE on q/k, scatter q/k (B,H,N,d) + vT (B,H,d,N))

__global__ void k_gemm(const u16* __restrict__ A, const u16* __restrict__ WT,
                       const float* __restrict__ bias, int K, int NC, int mode,
                       const float* __restrict__ resid, float* __restrict__ outF,
                       u16* __restrict__ outB,
                       const float* __restrict__ cosP, const float* __restrict__ sinP,
                       u16* __restrict__ qB, u16* __restrict__ kB, u16* __restrict__ vTB) {
    __shared__ __align__(16) u32 ldsA[32 * 512 / 2];   // up to 32 rows x K=512 bf16
    const int tid = threadIdx.x;
    const int lane = tid & 31, w = tid >> 5;
    const int m0 = blockIdx.y * 32;

    // async-DMA 32 x K bf16 A-tile into LDS (ASYNCcnt path)
    {
        const char* gbase = (const char*)A + (size_t)m0 * K * 2;
        u32 lbase = (u32)(uintptr_t)(&ldsA[0]);
        const int iters = K / 64;                      // (32*K*2) / (256*16)
        for (int i = 0; i < iters; ++i) {
            u32 boff = (u32)(tid + i * 256) * 16;
            async_g2l_b128(lbase + boff, gbase + boff);
        }
        wait_asynccnt0();
    }
    __syncthreads();

    const int rw = w >> 2;                 // row-half of block tile
    const int cw = w & 3;                  // col group
    const int r0 = rw * 16;
    const int col0 = blockIdx.x * 256 + cw * 64;
    const int rowA = r0 + (lane & 15);
    const int half8 = (lane >> 4) * 8;     // also the element-subtile offset
    const v4u* ldsA4 = (const v4u*)ldsA;
    const v4u* Wq = (const v4u*)WT;

    v8f acc[4];
    #pragma unroll
    for (int t = 0; t < 4; ++t)
        #pragma unroll
        for (int i = 0; i < 8; ++i) acc[t][i] = 0.0f;

    for (int kb = 0; kb < K; kb += 32) {
        BU au;
        ld_op_global(au, ldsA4, rowA * K + kb + half8);
        #pragma unroll
        for (int t = 0; t < 4; ++t) {
            int col = col0 + t * 16 + (lane & 15);
            __builtin_prefetch(((const char*)WT) + (size_t)(col * K + kb + 128) * 2, 0, 1);
            BU bu;
            ld_op_global(bu, Wq, col * K + kb + half8);
            acc[t] = wmma_bf16(au, bu, acc[t]);
        }
    }

    if (mode == 3) {
        #pragma unroll
        for (int t = 0; t < 4; ++t) {
            int col = col0 + t * 16 + (lane & 15);
            float bcol = bias[col];
            int region = col >> 8;         // 0=q 1=k 2=v
            int head = (col >> 5) & 7;
            int dim = col & 31;
            #pragma unroll
            for (int v = 0; v < 8; ++v) {
                int m = m0 + r0 + v + half8;
                int bb = m >> 11, np = m & 2047;
                float val = acc[t][v] + bcol;
                if (region < 2) {
                    float partner = __shfl_xor(val, 1, 32);  // adjacent dim pair
                    float cs = cosP[np * 16 + (dim >> 1)];
                    float sn = sinP[np * 16 + (dim >> 1)];
                    float r = ((dim & 1) == 0) ? (val * cs - partner * sn)
                                               : (partner * sn + val * cs);
                    u16* dst = (region == 0) ? qB : kB;
                    dst[(((bb * Hh + head) * Nseq) + np) * dh + dim] = f2bf(r);
                } else {
                    vTB[(((bb * Hh + head) * dh) + dim) * Nseq + np] = f2bf(val);
                }
            }
        }
    } else if (mode == 1) {
        #pragma unroll
        for (int t = 0; t < 4; ++t) {
            int col = col0 + t * 16 + (lane & 15);
            float bcol = bias[col];
            #pragma unroll
            for (int v = 0; v < 8; ++v) {
                int m = m0 + r0 + v + half8;
                outB[m * NC + col] = f2bf(fmaxf(acc[t][v] + bcol, 0.0f));
            }
        }
    } else {
        #pragma unroll
        for (int t = 0; t < 4; ++t) {
            int col = col0 + t * 16 + (lane & 15);
            float bcol = bias[col];
            #pragma unroll
            for (int v = 0; v < 8; ++v) {
                int m = m0 + r0 + v + half8;
                outF[m * NC + col] = resid[m * NC + col] + acc[t][v] + bcol;
            }
        }
    }
}

// ---- flash attention: 1 wave = 16 q rows; block = 128 rows of one (b,h) ---

__global__ void k_attn(const u16* __restrict__ qB, const u16* __restrict__ kB,
                       const u16* __restrict__ vTB, u16* __restrict__ attnB) {
    __shared__ __align__(16) u32 ldsP[8 * 256];  // per-wave 16x32 bf16 P patch
    const int tid = threadIdx.x, lane = tid & 31, w = tid >> 5;
    const int bh = blockIdx.y;
    const int bb = bh >> 3, h = bh & 7;
    const int q0 = blockIdx.x * 128 + w * 16;
    const int base = bh << 16;                   // *(N*d) in bf16 elements
    const int rowA = lane & 15;
    const int half8 = (lane >> 4) * 8;

    const v4u* qq = (const v4u*)qB;
    const v4u* kq = (const v4u*)kB;
    const v4u* vq = (const v4u*)vTB;

    // Q tile -> A layout (K dim = d = 32, exactly one WMMA K)
    BU aq;
    ld_op_global(aq, qq, base + (q0 + rowA) * dh + half8);

    v8f olo, ohi;
    float m8[8], l8[8];
    #pragma unroll
    for (int v = 0; v < 8; ++v) { olo[v] = 0.f; ohi[v] = 0.f; m8[v] = -3.0e38f; l8[v] = 0.f; }

    u16* ldsPs = (u16*)&ldsP[w * 256];
    const v4u* ldsPq = (const v4u*)&ldsP[w * 256];

    for (int j = 0; j < Nseq; j += 32) {
        // S = Q * K^T for 32 keys (two 16-key tiles)
        int key0 = j + (lane & 15);
        BU b0, b1;
        ld_op_global(b0, kq, base + key0 * dh + half8);
        ld_op_global(b1, kq, base + (key0 + 16) * dh + half8);
        v8f z;
        #pragma unroll
        for (int i = 0; i < 8; ++i) z[i] = 0.0f;
        v8f s0 = wmma_bf16(aq, b0, z);
        v8f s1 = wmma_bf16(aq, b1, z);

        // online softmax per q-row (row = v + half8; 16 lanes per row-group)
        #pragma unroll
        for (int v = 0; v < 8; ++v) {
            s0[v] *= SCALE; s1[v] *= SCALE;
            float rm = redmax16(fmaxf(s0[v], s1[v]));
            float nm = fmaxf(m8[v], rm);
            float alpha = __expf(m8[v] - nm);
            m8[v] = nm;
            s0[v] = __expf(s0[v] - nm);
            s1[v] = __expf(s1[v] - nm);
            float rs = redsum16(s0[v] + s1[v]);
            l8[v] = l8[v] * alpha + rs;
            olo[v] *= alpha; ohi[v] *= alpha;
        }

        // P: C-layout -> LDS (bf16) -> A-layout (per-wave patch, DS in-order)
        #pragma unroll
        for (int v = 0; v < 8; ++v) {
            int r = v + half8, cc = lane & 15;
            ldsPs[r * 32 + cc]      = f2bf(s0[v]);
            ldsPs[r * 32 + cc + 16] = f2bf(s1[v]);
        }
        BU pa;
        ld_op_global(pa, ldsPq, rowA * 32 + half8);

        // V tiles from transposed V: key pairs contiguous dwords
        BU blo, bhi;
        int dn = lane & 15;
        ld_op_global(blo, vq, base + dn * Nseq + j + half8);
        ld_op_global(bhi, vq, base + (dn + 16) * Nseq + j + half8);
        olo = wmma_bf16(pa, blo, olo);
        ohi = wmma_bf16(pa, bhi, ohi);
    }

    // normalize, scatter to (B,N,D) bf16
    #pragma unroll
    for (int v = 0; v < 8; ++v) {
        float inv = 1.0f / l8[v];
        int r = q0 + v + half8;
        int n = lane & 15;
        int rowoff = (bb * Nseq + r) * Dm + h * dh;
        attnB[rowoff + n]      = f2bf(olo[v] * inv);
        attnB[rowoff + n + 16] = f2bf(ohi[v] * inv);
    }
}

// ---- host orchestration ---------------------------------------------------

extern "C" void kernel_launch(void* const* d_in, const int* in_sizes, int n_in,
                              void* d_out, int out_size, void* d_ws, size_t ws_size,
                              hipStream_t stream) {
    const float* x      = (const float*)d_in[0];
    const float* fcos   = (const float*)d_in[1];
    const float* fsin   = (const float*)d_in[2];
    const float* w_qkv  = (const float*)d_in[3];
    const float* b_qkv  = (const float*)d_in[4];
    const float* w_proj = (const float*)d_in[5];
    const float* b_proj = (const float*)d_in[6];
    const float* ln1_g  = (const float*)d_in[7];
    const float* ln1_b  = (const float*)d_in[8];
    const float* ln2_g  = (const float*)d_in[9];
    const float* ln2_b  = (const float*)d_in[10];
    const float* w1     = (const float*)d_in[11];
    const float* b1     = (const float*)d_in[12];
    const float* w2     = (const float*)d_in[13];
    const float* b2     = (const float*)d_in[14];
    float* out = (float*)d_out;

    char* ws = (char*)d_ws;
    size_t off = 0;
    auto alloc = [&](size_t bytes) -> void* {
        void* p = ws + off;
        off += (bytes + 255) & ~(size_t)255;
        return p;
    };
    const int BN = Bsz * Nseq;                         // 8192
    u16*   hb     = (u16*)alloc((size_t)BN * Dm * 2);  // LN1 out
    u16*   h2     = (u16*)alloc((size_t)BN * Dm * 2);  // LN2 out
    u16*   attnb  = (u16*)alloc((size_t)BN * Dm * 2);  // attention out
    u16*   tmid   = (u16*)alloc((size_t)BN * 512 * 2); // FFN mid
    u16*   qb     = (u16*)alloc((size_t)BN * Dm * 2);
    u16*   kb     = (u16*)alloc((size_t)BN * Dm * 2);
    u16*   vtb    = (u16*)alloc((size_t)BN * Dm * 2);
    float* x1     = (float*)alloc((size_t)BN * Dm * 4);
    u16*   wqkvT  = (u16*)alloc((size_t)256 * 768 * 2);
    u16*   wprojT = (u16*)alloc((size_t)256 * 256 * 2);
    u16*   w1T    = (u16*)alloc((size_t)256 * 512 * 2);
    u16*   w2T    = (u16*)alloc((size_t)512 * 256 * 2);

    // 1) weights -> transposed bf16
    k_transpose_bf<<<(256 * 768 + 255) / 256, 256, 0, stream>>>(w_qkv,  wqkvT,  256, 768);
    k_transpose_bf<<<(256 * 256 + 255) / 256, 256, 0, stream>>>(w_proj, wprojT, 256, 256);
    k_transpose_bf<<<(256 * 512 + 255) / 256, 256, 0, stream>>>(w1,     w1T,    256, 512);
    k_transpose_bf<<<(512 * 256 + 255) / 256, 256, 0, stream>>>(w2,     w2T,    512, 256);

    // 2) LN1
    k_layernorm<<<BN, 256, 0, stream>>>(x, ln1_g, ln1_b, hb);

    // 3) QKV GEMM + RoPE + scatter (mode 3)
    k_gemm<<<dim3(768 / 256, BN / 32), 256, 0, stream>>>(
        hb, wqkvT, b_qkv, 256, 768, 3,
        nullptr, nullptr, nullptr, fcos, fsin, qb, kb, vtb);

    // 4) flash attention
    k_attn<<<dim3(Nseq / 128, Bsz * Hh), 256, 0, stream>>>(qb, kb, vtb, attnb);

    // 5) proj + residual -> x1 (mode 0)
    k_gemm<<<dim3(256 / 256, BN / 32), 256, 0, stream>>>(
        attnb, wprojT, b_proj, 256, 256, 0,
        x, x1, nullptr, nullptr, nullptr, nullptr, nullptr, nullptr);

    // 6) LN2
    k_layernorm<<<BN, 256, 0, stream>>>(x1, ln2_g, ln2_b, h2);

    // 7) FFN1 + ReLU (mode 1)
    k_gemm<<<dim3(512 / 256, BN / 32), 256, 0, stream>>>(
        h2, w1T, b1, 256, 512, 1,
        nullptr, nullptr, tmid, nullptr, nullptr, nullptr, nullptr, nullptr);

    // 8) FFN2 + residual -> out (mode 2)
    k_gemm<<<dim3(256 / 256, BN / 32), 256, 0, stream>>>(
        tmid, w2T, b2, 512, 256, 2,
        x1, out, nullptr, nullptr, nullptr, nullptr, nullptr, nullptr);
}
